// PointNet2MSG_16037407883889
// MI455X (gfx1250) — matrix operand order repaired and checked by reading, not compiled, not run on hardware
//
#include <hip/hip_runtime.h>
#include <hip/hip_bf16.h>
#include <math.h>

// ---------------------------------------------------------------------------
// Types for WMMA fragments (gfx1250, wave32)
// ---------------------------------------------------------------------------
typedef _Float16 h8v  __attribute__((ext_vector_type(8)));
typedef _Float16 h16v __attribute__((ext_vector_type(16)));
typedef float    f8v  __attribute__((ext_vector_type(8)));
typedef int      i4v  __attribute__((ext_vector_type(4)));

union ABFrag { h16v v; h8v h[2]; };

#if defined(__gfx1250__) && __has_builtin(__builtin_amdgcn_global_load_async_to_lds_b128) && __has_builtin(__builtin_amdgcn_s_wait_asynccnt)
#define USE_ASYNC_LDS 1
#else
#define USE_ASYNC_LDS 0
#endif

// ---------------------------------------------------------------------------
// Fold BN into weights:  w16[o,kpad] = w[o,i] * gamma/sqrt(var+eps), pad zeros
//                        b32[o]      = b*scale + beta - mean*scale
// ---------------------------------------------------------------------------
__global__ void fold_params_kernel(const float* __restrict__ w,
                                   const float* __restrict__ b,
                                   const float* __restrict__ gamma,
                                   const float* __restrict__ beta,
                                   const float* __restrict__ mean,
                                   const float* __restrict__ var,
                                   _Float16* __restrict__ w16,
                                   float* __restrict__ b32,
                                   int cin, int cout, int kpad, int fold) {
  int t = blockIdx.x * blockDim.x + threadIdx.x;
  if (t < cout) {
    float sc = fold ? gamma[t] * rsqrtf(var[t] + 1e-5f) : 1.0f;
    float sh = fold ? (beta[t] - mean[t] * sc) : 0.0f;
    b32[t] = b[t] * sc + sh;
  }
  int total = cout * kpad;
  if (t >= total) return;
  int o = t / kpad, i = t - o * kpad;
  float sc = fold ? gamma[o] * rsqrtf(var[o] + 1e-5f) : 1.0f;
  w16[t] = (_Float16)((i < cin) ? w[(size_t)o * cin + i] * sc : 0.0f);
}

// ---------------------------------------------------------------------------
// Farthest point sampling: one workgroup per batch, dist[] in LDS.
// ---------------------------------------------------------------------------
__global__ void fps_kernel(const float* __restrict__ xyz, int n, int stride,
                           int np, int* __restrict__ idx,
                           float* __restrict__ newxyz) {
  __shared__ float dist[4096];
  __shared__ float rv[256];
  __shared__ int   ri[256];
  __shared__ float cen[3];
  int b = blockIdx.x;
  int tid = threadIdx.x;
  const float* base = xyz + (size_t)b * n * stride;
  for (int i = tid; i < n; i += 256) dist[i] = 1e10f;
  __syncthreads();
  int far = 0;
  for (int it = 0; it < np; ++it) {
    if (tid == 0) {
      const float* c = base + (size_t)far * stride;
      cen[0] = c[0]; cen[1] = c[1]; cen[2] = c[2];
      idx[b * np + it] = far;
      float* o = newxyz + (size_t)(b * np + it) * 3;
      o[0] = c[0]; o[1] = c[1]; o[2] = c[2];
    }
    __syncthreads();
    float bv = -1.0f; int bi = n - 1;
    float cx = cen[0], cy = cen[1], cz = cen[2];
    for (int i = tid; i < n; i += 256) {
      const float* p = base + (size_t)i * stride;
      float dx = p[0] - cx, dy = p[1] - cy, dz = p[2] - cz;
      float d  = dx * dx + dy * dy + dz * dz;
      float dd = dist[i];
      dd = d < dd ? d : dd;
      dist[i] = dd;
      if (dd > bv || (dd == bv && i < bi)) { bv = dd; bi = i; }
    }
    rv[tid] = bv; ri[tid] = bi;
    __syncthreads();
    for (int s = 128; s > 0; s >>= 1) {
      if (tid < s) {
        if (rv[tid + s] > rv[tid] ||
            (rv[tid + s] == rv[tid] && ri[tid + s] < ri[tid])) {
          rv[tid] = rv[tid + s]; ri[tid] = ri[tid + s];
        }
      }
      __syncthreads();
    }
    far = ri[0];
    __syncthreads();
  }
}

// ---------------------------------------------------------------------------
// Ball query: one wave32 per centroid; ordered first-k selection via
// ballot_w32 + popc prefix sums (matches reference sort-then-truncate).
// ---------------------------------------------------------------------------
__global__ void ball_query_kernel(const float* __restrict__ xyz, int n, int stride,
                                  const float* __restrict__ newxyz, int np,
                                  float r2, int k, int* __restrict__ out) {
  int wpb = blockDim.x >> 5;
  int sg  = blockIdx.x * wpb + (threadIdx.x >> 5);
  int total = 8 * np;
  if (sg >= total) return;
  int b = sg / np, s = sg - b * np;
  int lane = threadIdx.x & 31;
  const float* base = xyz + (size_t)b * n * stride;
  const float* cen  = newxyz + (size_t)(b * np + s) * 3;
  float cx = cen[0], cy = cen[1], cz = cen[2];
  int* o = out + (size_t)(b * np + s) * k;
  int cnt = 0, first = -1;
  for (int i0 = 0; i0 < n && cnt < k; i0 += 32) {
    int i = i0 + lane;
    bool pred = false;
    if (i < n) {
      const float* p = base + (size_t)i * stride;
      float dx = p[0] - cx, dy = p[1] - cy, dz = p[2] - cz;
      pred = (dx * dx + dy * dy + dz * dz) <= r2;
    }
    unsigned mask = __builtin_amdgcn_ballot_w32(pred);
    if (first < 0 && mask) first = i0 + __ffs(mask) - 1;
    if (pred) {
      int pos = cnt + __popc(mask & ((1u << lane) - 1u));
      if (pos < k) o[pos] = i;
    }
    cnt += __popc(mask);
  }
  if (first < 0) first = 0;
  int c = cnt < k ? cnt : k;
  for (int j = c + lane; j < k; j += 32) o[j] = first;
}

// ---------------------------------------------------------------------------
// Gather kernels: build f16 feature rows [points | xyz - centroid | zeros]
// ---------------------------------------------------------------------------
__global__ void gather_sa1_kernel(const float* __restrict__ pc,
                                  const float* __restrict__ newxyz,
                                  const int* __restrict__ idx,
                                  _Float16* __restrict__ feat,
                                  int np, int k, int kpad) {
  int row = blockIdx.x * blockDim.x + threadIdx.x;
  int total = 8 * np * k;
  if (row >= total) return;
  int j = row % k, s = (row / k) % np, b = row / (k * np);
  int gi = idx[((size_t)b * np + s) * k + j];
  const float* src = pc + ((size_t)b * 4096 + gi) * 6;
  const float* cen = newxyz + ((size_t)b * np + s) * 3;
  _Float16* o = feat + (size_t)row * kpad;
  o[0] = (_Float16)src[3]; o[1] = (_Float16)src[4]; o[2] = (_Float16)src[5];
  o[3] = (_Float16)(src[0] - cen[0]);
  o[4] = (_Float16)(src[1] - cen[1]);
  o[5] = (_Float16)(src[2] - cen[2]);
  for (int cq = 6; cq < kpad; ++cq) o[cq] = (_Float16)0.0f;
}

__global__ void gather_sa2_kernel(const float* __restrict__ xyz,
                                  const _Float16* __restrict__ pts,
                                  const float* __restrict__ newxyz,
                                  const int* __restrict__ idx,
                                  _Float16* __restrict__ feat,
                                  int n, int np, int k, int cpts, int kpad) {
  int row = blockIdx.x * blockDim.x + threadIdx.x;
  int total = 8 * np * k;
  if (row >= total) return;
  int j = row % k, s = (row / k) % np, b = row / (k * np);
  int gi = idx[((size_t)b * np + s) * k + j];
  const _Float16* p = pts + ((size_t)b * n + gi) * cpts;
  _Float16* o = feat + (size_t)row * kpad;
  for (int cq = 0; cq < cpts; ++cq) o[cq] = p[cq];
  const float* q   = xyz + ((size_t)b * n + gi) * 3;
  const float* cen = newxyz + ((size_t)b * np + s) * 3;
  o[cpts + 0] = (_Float16)(q[0] - cen[0]);
  o[cpts + 1] = (_Float16)(q[1] - cen[1]);
  o[cpts + 2] = (_Float16)(q[2] - cen[2]);
  for (int cq = cpts + 3; cq < kpad; ++cq) o[cq] = (_Float16)0.0f;
}

// SA3 input: concat([l2_xyz, l2_points]) per point, padded to 672
__global__ void gather_sa3_kernel(const float* __restrict__ l2xyz,
                                  const _Float16* __restrict__ l2p,
                                  _Float16* __restrict__ feat) {
  int row = blockIdx.x * blockDim.x + threadIdx.x; // 8*128 rows
  if (row >= 8 * 128) return;
  _Float16* o = feat + (size_t)row * 672;
  const float* q = l2xyz + (size_t)row * 3;
  o[0] = (_Float16)q[0]; o[1] = (_Float16)q[1]; o[2] = (_Float16)q[2];
  const _Float16* p = l2p + (size_t)row * 640;
  for (int cq = 0; cq < 640; ++cq) o[3 + cq] = p[cq];
  for (int cq = 643; cq < 672; ++cq) o[cq] = (_Float16)0.0f;
}

// ---------------------------------------------------------------------------
// WMMA GEMM: Y[M,Cout] = relu(X[M,Kpad] * W[Cout,Kpad]^T + bias)
// Block = 4 waves over M (16 rows each); block N-tile = 32 cols.
// Each wave holds 2 f32 accumulators (16x32 output): A fragment reused 2x,
// two independent WMMAs back-to-back per k-step.
// Weight chunk (32 cout rows x 32 K-halves) staged to LDS per k-step via
// CDNA5 async global->LDS (ASYNCcnt) when available, then read back with
// ds_load_b128 using a 40-half pitch (conflict-free 4-bank groups/lane).
// Exact grid => EXEC all-ones, as WMMA requires.
// ---------------------------------------------------------------------------
__global__ void wmma_gemm_kernel(const _Float16* __restrict__ X,
                                 const _Float16* __restrict__ W,
                                 const float* __restrict__ bias,
                                 _Float16* __restrict__ Y,
                                 int M, int Kpad, int Cout) {
  __shared__ alignas(16) _Float16 lB[32 * 40]; // 32 rows, pitch 40 halves
  int lane = threadIdx.x & 31;
  int wave = threadIdx.x >> 5;
  int mbase = (blockIdx.x * 4 + wave) * 16;
  int nbase = blockIdx.y * 32;
  int mrow  = mbase + (lane & 15);
  int aoff  = (lane >> 4) * 16;
  int boff  = (lane >> 4) * 16;
  int nl    = lane & 15;
  const _Float16* xrow = X + (size_t)mrow * Kpad;
  // staging map: each of the 128 threads copies one b128 chunk per k-step
  int sr = threadIdx.x >> 2;        // cout row 0..31
  int sc = (threadIdx.x & 3) * 8;   // half offset 0/8/16/24
  const _Float16* wsrc = W + (size_t)(nbase + sr) * Kpad + sc;
  _Float16* wdst = lB + sr * 40 + sc;
  f8v c0 = {0.f, 0.f, 0.f, 0.f, 0.f, 0.f, 0.f, 0.f};
  f8v c1 = {0.f, 0.f, 0.f, 0.f, 0.f, 0.f, 0.f, 0.f};
  for (int k0 = 0; k0 < Kpad; k0 += 32) {
#if USE_ASYNC_LDS
    {
      void* gsrc = (void*)(wsrc + k0);
      void* ldst = (void*)wdst;
      __builtin_amdgcn_global_load_async_to_lds_b128(
          (__attribute__((address_space(1))) i4v*)gsrc,
          (__attribute__((address_space(3))) i4v*)ldst, 0, 0);
      __builtin_amdgcn_s_wait_asynccnt(0);
    }
#else
    *reinterpret_cast<h8v*>(wdst) = *reinterpret_cast<const h8v*>(wsrc + k0);
#endif
    __syncthreads();
    ABFrag a, b0, b1;
    a.h[0] = *reinterpret_cast<const h8v*>(xrow + k0 + aoff);
    a.h[1] = *reinterpret_cast<const h8v*>(xrow + k0 + aoff + 8);
    b0.h[0] = *reinterpret_cast<const h8v*>(lB + nl * 40 + boff);
    b0.h[1] = *reinterpret_cast<const h8v*>(lB + nl * 40 + boff + 8);
    b1.h[0] = *reinterpret_cast<const h8v*>(lB + (nl + 16) * 40 + boff);
    b1.h[1] = *reinterpret_cast<const h8v*>(lB + (nl + 16) * 40 + boff + 8);
    if (k0 + 32 < Kpad) __builtin_prefetch(xrow + k0 + 32, 0, 0);
    c0 = __builtin_amdgcn_wmma_f32_16x16x32_f16(false, a.v, false, b0.v,
                                                (short)0, c0, false, false);
    c1 = __builtin_amdgcn_wmma_f32_16x16x32_f16(false, a.v, false, b1.v,
                                                (short)0, c1, false, false);
    __syncthreads();
  }
  int col0 = nbase + nl;
  int col1 = nbase + 16 + nl;
  float bb0 = bias[col0];
  float bb1 = bias[col1];
  int rbase = mbase + ((lane >> 4) ? 8 : 0);
#pragma unroll
  for (int r = 0; r < 8; ++r) {
    float v0 = c0[r] + bb0;
    float v1 = c1[r] + bb1;
    v0 = v0 > 0.f ? v0 : 0.f;
    v1 = v1 > 0.f ? v1 : 0.f;
    Y[(size_t)(rbase + r) * Cout + col0] = (_Float16)v0;
    Y[(size_t)(rbase + r) * Cout + col1] = (_Float16)v1;
  }
}

// ---------------------------------------------------------------------------
// Max over the k neighbor dimension. Optional f16 dest (with stride/offset)
// and up to two f32 dests (L3 buffer + d_out view for SA3).
// ---------------------------------------------------------------------------
__global__ void max_over_k_kernel(const _Float16* __restrict__ X, int rows,
                                  int k, int C, _Float16* __restrict__ outH,
                                  float* __restrict__ outF1,
                                  float* __restrict__ outF2,
                                  int dstStride, int colOff) {
  int t = blockIdx.x * blockDim.x + threadIdx.x;
  int total = rows * C;
  if (t >= total) return;
  int r = t / C, cq = t - r * C;
  const _Float16* p = X + ((size_t)r * k) * C + cq;
  float m = -1e30f;
  for (int j = 0; j < k; ++j) {
    float v = (float)p[(size_t)j * C];
    m = v > m ? v : m;
  }
  if (outH)  outH[(size_t)r * dstStride + colOff + cq] = (_Float16)m;
  if (outF1) outF1[(size_t)r * dstStride + colOff + cq] = m;
  if (outF2) outF2[(size_t)r * dstStride + colOff + cq] = m;
}

// ---------------------------------------------------------------------------
// Small FC layers (M=8): y = [relu]( x . w16^T + bias )
// ---------------------------------------------------------------------------
__global__ void fc_kernel(const float* __restrict__ x,
                          const _Float16* __restrict__ w,
                          const float* __restrict__ bias,
                          float* __restrict__ y,
                          int Bn, int cin, int kpad, int cout, int relu) {
  int t = blockIdx.x * blockDim.x + threadIdx.x;
  if (t >= Bn * cout) return;
  int b = t / cout, o = t - b * cout;
  const float* xr = x + (size_t)b * cin;
  const _Float16* wr = w + (size_t)o * kpad;
  float acc = bias[o];
  for (int i = 0; i < cin; ++i) acc += xr[i] * (float)wr[i];
  if (relu) acc = acc > 0.f ? acc : 0.f;
  y[t] = acc;
}

__global__ void log_softmax_kernel(const float* __restrict__ logits,
                                   float* __restrict__ out) {
  int b = threadIdx.x;
  if (b >= 8) return;
  const float* r = logits + b * 16;
  float m = -1e30f;
  for (int i = 0; i < 16; ++i) m = r[i] > m ? r[i] : m;
  float s = 0.f;
  for (int i = 0; i < 16; ++i) s += expf(r[i] - m);
  float ls = logf(s);
  for (int i = 0; i < 16; ++i) out[b * 16 + i] = r[i] - m - ls;
}

// ---------------------------------------------------------------------------
// Host orchestration
// ---------------------------------------------------------------------------
static const int L_CIN[24]  = {6,32,32, 6,64,64, 6,64,96,
                               323,64,64, 323,128,128, 323,128,128,
                               643,256,512, 1024,512,256};
static const int L_COUT[24] = {32,32,64, 64,64,128, 64,96,128,
                               64,64,128, 128,128,256, 128,128,256,
                               256,512,1024, 512,256,16};
// sorted-key pytree flatten (params first: fc1,fc2,fc3,sa1,sa2,sa3; pc last)
static const int BASE_SORT[24] = {18,24,30, 36,42,48, 54,60,66,
                                  72,78,84, 90,96,102, 108,114,120,
                                  126,132,138, 0,6,12};
// insertion-order flatten (pc first, then sa1,sa2,sa3,fc1,fc2,fc3)
static const int BASE_INS[24]  = {1,7,13, 19,25,31, 37,43,49,
                                  55,61,67, 73,79,85, 91,97,103,
                                  109,115,121, 127,133,139};

static inline int kpad_of(int cin) { return (cin + 31) & ~31; }

extern "C" void kernel_launch(void* const* d_in, const int* in_sizes, int n_in,
                              void* d_out, int out_size, void* d_ws, size_t ws_size,
                              hipStream_t stream) {
  char* ws = (char*)d_ws;
  // ---- workspace layout ----
  size_t off = 0;
  auto alloc = [&](size_t bytes) {
    size_t o = off; off += (bytes + 255) & ~(size_t)255; return o;
  };
  size_t FEATA = alloc((size_t)524288 * 128 * 2);
  size_t FEATB = alloc((size_t)524288 * 128 * 2);
  size_t IDX   = alloc((size_t)8 * 512 * 128 * 4);
  size_t FPSI  = alloc((size_t)8 * 512 * 4);
  size_t NX1   = alloc((size_t)8 * 512 * 3 * 4);
  size_t L1P   = alloc((size_t)8 * 512 * 320 * 2);
  size_t NX2   = alloc((size_t)8 * 128 * 3 * 4);
  size_t L2P   = alloc((size_t)8 * 128 * 640 * 2);
  size_t L3    = alloc((size_t)8 * 1024 * 4);
  size_t H1    = alloc((size_t)8 * 512 * 4);
  size_t H2    = alloc((size_t)8 * 256 * 4);
  size_t LOGI  = alloc((size_t)8 * 16 * 4);
  size_t BIAS  = alloc((size_t)24 * 1024 * 4);
  size_t WOFF[24];
  for (int l = 0; l < 24; ++l)
    WOFF[l] = alloc((size_t)L_COUT[l] * kpad_of(L_CIN[l]) * 2);
  (void)ws_size; (void)n_in; (void)out_size;

  // ---- input mapping (detect flatten convention by pc size) ----
  int pcFirst = (in_sizes[0] == 8 * 4096 * 6) ? 1 : 0;
  const float* pc = (const float*)d_in[pcFirst ? 0 : 144];
  auto F = [&](int i) { return (const float*)d_in[i]; };

  // ---- fold BN into f16 weights ----
  for (int l = 0; l < 24; ++l) {
    int base = pcFirst ? BASE_INS[l] : BASE_SORT[l];
    const float *w, *b, *g, *be, *me, *va;
    if (pcFirst) { // insertion order within layer: w,b,gamma,beta,mean,var
      w = F(base + 0); b = F(base + 1); g = F(base + 2);
      be = F(base + 3); me = F(base + 4); va = F(base + 5);
    } else {       // sorted: b,beta,gamma,mean,var,w
      b = F(base + 0); be = F(base + 1); g = F(base + 2);
      me = F(base + 3); va = F(base + 4); w = F(base + 5);
    }
    int kp = kpad_of(L_CIN[l]);
    int tot = L_COUT[l] * kp;
    fold_params_kernel<<<(tot + 255) / 256, 256, 0, stream>>>(
        w, b, g, be, me, va, (_Float16*)(ws + WOFF[l]),
        (float*)(ws + BIAS) + l * 1024, L_CIN[l], L_COUT[l], kp, l < 23 ? 1 : 0);
  }

  _Float16* fa = (_Float16*)(ws + FEATA);
  _Float16* fb = (_Float16*)(ws + FEATB);

  auto run_mlp = [&](int l0, int M) -> _Float16* {
    _Float16* cur = fa; _Float16* nxt = fb;
    for (int li = 0; li < 3; ++li) {
      int l = l0 + li;
      dim3 g(M / 64, L_COUT[l] / 32);
      wmma_gemm_kernel<<<g, 128, 0, stream>>>(
          cur, (const _Float16*)(ws + WOFF[l]),
          (const float*)(ws + BIAS) + l * 1024, nxt, M, kpad_of(L_CIN[l]),
          L_COUT[l]);
      _Float16* t = cur; cur = nxt; nxt = t;
    }
    return cur;
  };

  // ============================ SA1 ============================
  fps_kernel<<<8, 256, 0, stream>>>(pc, 4096, 6, 512, (int*)(ws + FPSI),
                                    (float*)(ws + NX1));
  {
    const float r1[3] = {0.1f, 0.2f, 0.4f};
    const int   k1[3] = {16, 32, 128};
    int colOff = 0;
    for (int br = 0; br < 3; ++br) {
      int k = k1[br];
      ball_query_kernel<<<(8 * 512) / 8, 256, 0, stream>>>(
          pc, 4096, 6, (const float*)(ws + NX1), 512, r1[br] * r1[br], k,
          (int*)(ws + IDX));
      int rows = 8 * 512 * k;
      gather_sa1_kernel<<<(rows + 255) / 256, 256, 0, stream>>>(
          pc, (const float*)(ws + NX1), (const int*)(ws + IDX), fa, 512, k, 32);
      _Float16* outb = run_mlp(br * 3, rows);
      int cl = L_COUT[br * 3 + 2];
      int tot = 8 * 512 * cl;
      max_over_k_kernel<<<(tot + 255) / 256, 256, 0, stream>>>(
          outb, 8 * 512, k, cl, (_Float16*)(ws + L1P), nullptr, nullptr, 320,
          colOff);
      colOff += cl; // 64, 128, 128
    }
  }

  // ============================ SA2 ============================
  fps_kernel<<<8, 256, 0, stream>>>((const float*)(ws + NX1), 512, 3, 128,
                                    (int*)(ws + FPSI), (float*)(ws + NX2));
  {
    const float r2[3] = {0.2f, 0.4f, 0.8f};
    const int   k2[3] = {32, 64, 128};
    int colOff = 0;
    for (int br = 0; br < 3; ++br) {
      int k = k2[br];
      ball_query_kernel<<<(8 * 128) / 8, 256, 0, stream>>>(
          (const float*)(ws + NX1), 512, 3, (const float*)(ws + NX2), 128,
          r2[br] * r2[br], k, (int*)(ws + IDX));
      int rows = 8 * 128 * k;
      gather_sa2_kernel<<<(rows + 255) / 256, 256, 0, stream>>>(
          (const float*)(ws + NX1), (const _Float16*)(ws + L1P),
          (const float*)(ws + NX2), (const int*)(ws + IDX), fa, 512, 128, k,
          320, 352);
      _Float16* outb = run_mlp(9 + br * 3, rows);
      int cl = L_COUT[9 + br * 3 + 2];
      int tot = 8 * 128 * cl;
      max_over_k_kernel<<<(tot + 255) / 256, 256, 0, stream>>>(
          outb, 8 * 128, k, cl, (_Float16*)(ws + L2P), nullptr, nullptr, 640,
          colOff);
      colOff += cl; // 128, 256, 256
    }
  }

  // ============================ SA3 ============================
  gather_sa3_kernel<<<(8 * 128 + 255) / 256, 256, 0, stream>>>(
      (const float*)(ws + NX2), (const _Float16*)(ws + L2P), fa);
  {
    _Float16* outb = run_mlp(18, 8 * 128);
    int tot = 8 * 1024;
    max_over_k_kernel<<<(tot + 255) / 256, 256, 0, stream>>>(
        outb, 8, 128, 1024, nullptr, (float*)(ws + L3), (float*)d_out + 128,
        1024, 0);
  }

  // ============================ FC head ============================
  fc_kernel<<<(8 * 512 + 255) / 256, 256, 0, stream>>>(
      (const float*)(ws + L3), (const _Float16*)(ws + WOFF[21]),
      (const float*)(ws + BIAS) + 21 * 1024, (float*)(ws + H1), 8, 1024, 1024,
      512, 1);
  fc_kernel<<<(8 * 256 + 255) / 256, 256, 0, stream>>>(
      (const float*)(ws + H1), (const _Float16*)(ws + WOFF[22]),
      (const float*)(ws + BIAS) + 22 * 1024, (float*)(ws + H2), 8, 512, 512,
      256, 1);
  fc_kernel<<<1, 256, 0, stream>>>(
      (const float*)(ws + H2), (const _Float16*)(ws + WOFF[23]),
      (const float*)(ws + BIAS) + 23 * 1024, (float*)(ws + LOGI), 8, 256, 256,
      16, 0);
  log_softmax_kernel<<<1, 32, 0, stream>>>((const float*)(ws + LOGI),
                                           (float*)d_out);
}
